// ImplicitWarpModule_38637525795037
// MI455X (gfx1250) — matrix operand order repaired and checked by reading, not compiled, not run on hardware
//
#include <hip/hip_runtime.h>
#include <hip/hip_bf16.h>
#include <math.h>

typedef __attribute__((ext_vector_type(16))) _Float16 v16h;
typedef __attribute__((ext_vector_type(8)))  float    v8f;

#define HH 192
#define WW 192
#define HWC (HH*WW)
#define CIN 48
#define KPAD 64
#define NDIM 128
#define NHEADS 8
#define HD 16
#define PXB 128            // pixels per block
#define NTHREADS 256       // 8 wave32s
#define ATT_SCALE 0.25f    // HEAD_DIM^-0.5 = 16^-0.5

__device__ __forceinline__ v8f wmma_f16(v16h a, v16h b, v8f c) {
  // (neg_a, A, neg_b, B, c_mod, C, reuse_a, reuse_b)
  return __builtin_amdgcn_wmma_f32_16x16x32_f16(false, a, false, b, (short)0, c, false, false);
}

// A-matrix fragment (16x32, M x K), row-major LDS [row][KPAD] f16.
// Layout (CDNA5 ISA): lane%16 = M; vgpr r holds K pair
//   k = (r<4 ? 2r : 8+2r) + (lane>=16 ? 8 : 0), elements (k, k+1).
__device__ __forceinline__ v16h load_fragA(const _Float16* base, int row, int k0, int hi) {
  union { v16h h; unsigned u[8]; } f;
#pragma unroll
  for (int r = 0; r < 8; ++r) {
    int kk = k0 + ((r < 4) ? 2*r : 8 + 2*r) + (hi ? 8 : 0);
    f.u[r] = *reinterpret_cast<const unsigned*>(base + row*KPAD + kk);
  }
  return f.h;
}

// B-matrix fragment (32x16, K x N), weights stored transposed [n][KPAD] f16.
// Layout: lane%16 = N; vgpr r holds K pair k = (r<4 ? 2r : 8+2r) + (lane>=16 ? 8 : 0) (same as A).
__device__ __forceinline__ v16h load_fragB(const _Float16* base, int col, int k0, int hi) {
  union { v16h h; unsigned u[8]; } f;
#pragma unroll
  for (int r = 0; r < 8; ++r) {
    int kk = k0 + ((r < 4) ? 2*r : 8 + 2*r) + (hi ? 8 : 0);
    f.u[r] = *reinterpret_cast<const unsigned*>(base + col*KPAD + kk);
  }
  return f.h;
}

// sine position-encoding channel: c<24 uses y-arg, c>=24 uses x-arg;
// dt[i] = 10000^((i//2)/12); even c -> sin, odd c -> cos.
__device__ __forceinline__ float pe_chan(int c, float ay, float ax) {
  int cm = (c < 24) ? c : c - 24;
  float inv_dt = __expf(-0.76752836433134865f * (float)(cm >> 1)); // ln(1e4)/12
  float arg = ((c < 24) ? ay : ax) * inv_dt;
  return (c & 1) ? __cosf(arg) : __sinf(arg);
}

__global__ void __launch_bounds__(NTHREADS, 1) iwarp_fused_kernel(
    const float* __restrict__ y, const float* __restrict__ x,
    const float* __restrict__ conv_w, const float* __restrict__ conv_b,
    const float* __restrict__ q_w, const float* __restrict__ q_b,
    const float* __restrict__ k_w, const float* __restrict__ k_b,
    const float* __restrict__ v_w, const float* __restrict__ v_b,
    float* __restrict__ out) {
  extern __shared__ char smraw[];
  _Float16* WqT = (_Float16*)smraw;              // [128][64] f16 (W^T, K padded)
  _Float16* WkT = WqT + NDIM*KPAD;
  _Float16* WvT = WkT + NDIM*KPAD;
  _Float16* Aq  = WvT + NDIM*KPAD;               // [PXB][64]
  _Float16* Ak  = Aq  + PXB*KPAD;                // [4][PXB][64]
  float*    pb  = (float*)(Ak + 4*PXB*KPAD);     // [4][48]

  const int tid = threadIdx.x;
  const int nb  = blockIdx.y;
  const int pbase = blockIdx.x * PXB;
  const float TWO_PI = 6.283185307179586f;

  // ---- Stage 0: weights -> LDS (transposed, f16, K-pair packed b32 stores) ----
  for (int idx = tid; idx < NDIM*(KPAD/2); idx += NTHREADS) {
    int nn = idx >> 5, kk = (idx & 31) << 1;     // kk even; CIN is even
    union { unsigned u; _Float16 h[2]; } tq, tk, tv;
    tq.u = 0u; tk.u = 0u; tv.u = 0u;
    if (kk < CIN) {
      tq.h[0] = (_Float16)q_w[kk*NDIM + nn];     tq.h[1] = (_Float16)q_w[(kk+1)*NDIM + nn];
      tk.h[0] = (_Float16)k_w[kk*NDIM + nn];     tk.h[1] = (_Float16)k_w[(kk+1)*NDIM + nn];
      tv.h[0] = (_Float16)v_w[kk*NDIM + nn];     tv.h[1] = (_Float16)v_w[(kk+1)*NDIM + nn];
    }
    *reinterpret_cast<unsigned*>(WqT + nn*KPAD + kk) = tq.u;
    *reinterpret_cast<unsigned*>(WkT + nn*KPAD + kk) = tk.u;
    *reinterpret_cast<unsigned*>(WvT + nn*KPAD + kk) = tv.u;
  }
  // window position bias pb[v][c]: (wy,wx) = (v>>1, v&1); ye=wy*2pi/(1+eps)
  if (tid < 4*CIN) {
    int vwin = tid / CIN, cc = tid - vwin*CIN;
    float ay = (float)(vwin >> 1) * (TWO_PI / (1.0f + 1e-6f));
    float axc = (float)(vwin & 1) * (TWO_PI / (1.0f + 1e-6f));
    pb[tid] = pe_chan(cc, ay, axc);
  }
  __syncthreads();

  // ---- Stage 1: one thread per pixel: conv offset, PE, gather -> LDS ----
  if (tid < PXB) {
    const int p = pbase + tid;
    const int i = p / WW, j = p - i*WW;
    const float* ybase = y + (size_t)nb * CIN * HWC;
    const float* xbase = x + (size_t)nb * CIN * HWC;

    float acc = conv_b[0];
    for (int ci = 0; ci < 2*CIN; ++ci) {
      const float* src = (ci < CIN) ? (ybase + (size_t)ci*HWC)
                                    : (xbase + (size_t)(ci-CIN)*HWC);
      const float* wr = conv_w + ci*9;
#pragma unroll
      for (int dy = -1; dy <= 1; ++dy) {
        int ii = i + dy;
        if (ii < 0 || ii >= HH) continue;
#pragma unroll
        for (int dx = -1; dx <= 1; ++dx) {
          int jj = j + dx;
          if (jj < 0 || jj >= WW) continue;
          acc += src[ii*WW + jj] * wr[(dy+1)*3 + (dx+1)];
        }
      }
    }
    float off   = acc * (2.0f / (float)WW);
    float fx    = (float)j + off;
    float flx   = floorf(fx);
    float fracx = fx - flx;
    float ax    = fracx * (TWO_PI / (2.0f + 1e-6f)); // frac_y == 0 always

    // x_in = x + points PE -> Aq row (f16 pairs, K-padded)
    for (int cc = 0; cc < CIN; cc += 2) {
      union { unsigned u; _Float16 h[2]; } t;
      t.h[0] = (_Float16)(xbase[(size_t)cc*HWC + p]     + pe_chan(cc,   0.0f, ax));
      t.h[1] = (_Float16)(xbase[(size_t)(cc+1)*HWC + p] + pe_chan(cc+1, 0.0f, ax));
      *reinterpret_cast<unsigned*>(Aq + tid*KPAD + cc) = t.u;
    }
    for (int cc = CIN; cc < KPAD; cc += 2)
      *reinterpret_cast<unsigned*>(Aq + tid*KPAD + cc) = 0u;

    // wrp rows: gather y at 4 clipped window positions + pb
    int ifl = (int)flx;
    for (int vwin = 0; vwin < 4; ++vwin) {
      int iy = i + (vwin >> 1);
      iy = iy < 0 ? 0 : (iy > HH-1 ? HH-1 : iy);
      int ix = ifl + (vwin & 1);
      ix = ix < 0 ? 0 : (ix > WW-1 ? WW-1 : ix);
      int lin = iy*WW + ix;
      _Float16* dst = Ak + ((size_t)vwin*PXB + tid)*KPAD;
      for (int cc = 0; cc < CIN; cc += 2) {
        union { unsigned u; _Float16 h[2]; } t;
        t.h[0] = (_Float16)(ybase[(size_t)cc*HWC + lin]     + pb[vwin*CIN + cc]);
        t.h[1] = (_Float16)(ybase[(size_t)(cc+1)*HWC + lin] + pb[vwin*CIN + cc + 1]);
        *reinterpret_cast<unsigned*>(dst + cc) = t.u;
      }
      for (int cc = CIN; cc < KPAD; cc += 2)
        *reinterpret_cast<unsigned*>(dst + cc) = 0u;
    }
  }
  __syncthreads();

  // ---- Stage 2: per-wave (16 pixels) WMMA GEMMs + in-fragment attention ----
  const int wv   = tid >> 5;
  const int lane = tid & 31;
  const int lm   = lane & 15;
  const int hi   = lane >> 4;
  const int arow = wv*16 + lm;
  float* outb = out + (size_t)nb * NDIM * HWC;

  // cache all A fragments in registers (launch_bounds gives us the VGPRs)
  v16h aq0 = load_fragA(Aq, arow, 0, hi);
  v16h aq1 = load_fragA(Aq, arow, 32, hi);
  v16h ak0[4], ak1[4];
#pragma unroll
  for (int jw = 0; jw < 4; ++jw) {
    const _Float16* akb = Ak + (size_t)jw*PXB*KPAD;
    ak0[jw] = load_fragA(akb, arow, 0, hi);
    ak1[jw] = load_fragA(akb, arow, 32, hi);
  }
  // preload per-head bias scalars (col = hh*16 + lane%16)
  float qb[NHEADS], kb[NHEADS], vb[NHEADS];
#pragma unroll
  for (int hh = 0; hh < NHEADS; ++hh) {
    qb[hh] = q_b[hh*HD + lm];
    kb[hh] = k_b[hh*HD + lm];
    vb[hh] = v_b[hh*HD + lm];
  }

  for (int hh = 0; hh < NHEADS; ++hh) {
    const int ncol = hh*HD + lm;   // WMMA N-tile column == head channel

    v16h bq0 = load_fragB(WqT, ncol, 0, hi), bq1 = load_fragB(WqT, ncol, 32, hi);
    v8f qacc = {};
    qacc = wmma_f16(aq0, bq0, qacc);
    qacc = wmma_f16(aq1, bq1, qacc);
#pragma unroll
    for (int r = 0; r < 8; ++r) qacc[r] += qb[hh];

    v16h bk0 = load_fragB(WkT, ncol, 0, hi), bk1 = load_fragB(WkT, ncol, 32, hi);
    float lg[8][4];
#pragma unroll
    for (int jw = 0; jw < 4; ++jw) {
      v8f kacc = {};
      kacc = wmma_f16(ak0[jw], bk0, kacc);
      kacc = wmma_f16(ak1[jw], bk1, kacc);
#pragma unroll
      for (int r = 0; r < 8; ++r) {
        // C/D layout: vgpr r, half hi -> row r+8*hi; lane%16 -> col.
        float pr = qacc[r] * (kacc[r] + kb[hh]);
        pr += __shfl_xor(pr, 1, 32);
        pr += __shfl_xor(pr, 2, 32);
        pr += __shfl_xor(pr, 4, 32);
        pr += __shfl_xor(pr, 8, 32);   // sum over 16-lane half = dot over head dim
        lg[r][jw] = pr * ATT_SCALE;
      }
    }
    // softmax over the 4 window keys, per row (replicated across the half)
#pragma unroll
    for (int r = 0; r < 8; ++r) {
      float m = fmaxf(fmaxf(lg[r][0], lg[r][1]), fmaxf(lg[r][2], lg[r][3]));
      float s = 0.f;
#pragma unroll
      for (int jw = 0; jw < 4; ++jw) { lg[r][jw] = __expf(lg[r][jw] - m); s += lg[r][jw]; }
      float inv = 1.0f / s;
#pragma unroll
      for (int jw = 0; jw < 4; ++jw) lg[r][jw] *= inv;
    }

    v16h bv0 = load_fragB(WvT, ncol, 0, hi), bv1 = load_fragB(WvT, ncol, 32, hi);
    v8f oacc = {};
#pragma unroll
    for (int jw = 0; jw < 4; ++jw) {
      v8f vacc = {};
      vacc = wmma_f16(ak0[jw], bv0, vacc);
      vacc = wmma_f16(ak1[jw], bv1, vacc);
#pragma unroll
      for (int r = 0; r < 8; ++r) oacc[r] += lg[r][jw] * (vacc[r] + vb[hh]);
    }

    // out[n][ncol][p], p = pbase + wv*16 + r + 8*hi : 8 contiguous floats/lane
    float* op = outb + (size_t)ncol * HWC + pbase + wv*16 + 8*hi;
#pragma unroll
    for (int r = 0; r < 8; ++r) op[r] = oacc[r];
  }
}

extern "C" void kernel_launch(void* const* d_in, const int* in_sizes, int n_in,
                              void* d_out, int out_size, void* d_ws, size_t ws_size,
                              hipStream_t stream) {
  (void)in_sizes; (void)n_in; (void)out_size; (void)d_ws; (void)ws_size;
  const float* y      = (const float*)d_in[0];
  const float* x      = (const float*)d_in[1];
  const float* conv_w = (const float*)d_in[2];
  const float* conv_b = (const float*)d_in[3];
  const float* q_w    = (const float*)d_in[4];
  const float* q_b    = (const float*)d_in[5];
  const float* k_w    = (const float*)d_in[6];
  const float* k_b    = (const float*)d_in[7];
  const float* v_w    = (const float*)d_in[8];
  const float* v_b    = (const float*)d_in[9];
  float* out = (float*)d_out;

  const size_t smem = (size_t)(3*NDIM*KPAD + PXB*KPAD + 4*PXB*KPAD) * sizeof(_Float16)
                    + (size_t)(4*CIN) * sizeof(float);   // ~129 KB (WGP has 320 KB)
  (void)hipFuncSetAttribute((const void*)iwarp_fused_kernel,
                            hipFuncAttributeMaxDynamicSharedMemorySize, (int)smem);

  dim3 grid(HWC / PXB, 2);   // 288 pixel-blocks x n=2
  dim3 block(NTHREADS);
  iwarp_fused_kernel<<<grid, block, smem, stream>>>(
      y, x, conv_w, conv_b, q_w, q_b, k_w, k_b, v_w, v_b, out);
}
